// DeepGMR_71012989272629
// MI455X (gfx1250) — matrix operand order repaired and checked by reading, not compiled, not run on hardware
//
#include <hip/hip_runtime.h>
#include <hip/hip_bf16.h>
#include <cstdint>

// ---------------------------------------------------------------------------
// DeepGMR forward for MI455X (gfx1250): WMMA f16 GEMM backbone + fused BN/ReLU
//   - v_wmma_f32_16x16x32_f16 compute
//   - W tiles staged with global_load_async_to_lds_b128 (ASYNCcnt path)
//   - X tiles staged through VGPRs with BN+ReLU of previous layer folded in
// ---------------------------------------------------------------------------

typedef _Float16 half_t;
typedef _Float16 v16h __attribute__((ext_vector_type(16)));
typedef float    v8f  __attribute__((ext_vector_type(8)));

#define BB          4
#define NPTS        16384
#define PTOT        65536      // B * N
#define DIN         80
#define DINP        96         // padded to multiple of 32 for K loop
#define FEAT_STRIDE 83         // 3 xyz + 80 features
#define OUT_ROW     49184      // 49152 transformed + 16 T_est + 16 T_inv

// ---------------------------------------------------------------------------
// Weight f32 -> f16 (with optional K padding, optional wider source stride)
// ---------------------------------------------------------------------------
__global__ void cvt_w(const float* __restrict__ src, half_t* __restrict__ dst,
                      int O, int I, int Ipad, int srcStride)
{
    int idx = blockIdx.x * 256 + threadIdx.x;
    if (idx >= O * Ipad) return;
    int o = idx / Ipad, i = idx - o * Ipad;
    dst[idx] = (i < I) ? (half_t)src[(size_t)o * srcStride + i] : (half_t)0.f;
}

// Pack input features (B,N,83)[...,3:] -> [P][96] f16 (zero padded)
__global__ void pack_input(const float* __restrict__ inp, half_t* __restrict__ X)
{
    int idx = blockIdx.x * 256 + threadIdx.x;
    if (idx >= PTOT * DINP) return;
    int p = idx / DINP, i = idx - p * DINP;
    X[idx] = (i < DIN) ? (half_t)inp[(size_t)p * FEAT_STRIDE + 3 + i] : (half_t)0.f;
}

__global__ void zero_f(float* a, int n)
{
    int i = blockIdx.x * 256 + threadIdx.x;
    if (i < n) a[i] = 0.f;
}

// ---------------------------------------------------------------------------
// Core GEMM: Y[P][O] = X[P][I] (affine+relu folded) * W[O][I]^T
//   - WMMA f32_16x16x32_f16, block tile 128 points x 64 out-ch, 8 waves
//   - W tile: async global->LDS copy (one b128 lane-copy per thread per step)
//   - epilogue: raw f16 store + per-channel sum/sumsq (LDS ds_add -> global)
// ---------------------------------------------------------------------------
template<bool HAS_ST, bool HAS_GB>
__global__ __launch_bounds__(256) void gemm_bn_wmma(
    const half_t* __restrict__ W,      // [O][I] f16
    const half_t* __restrict__ X,      // [P][I] f16 raw prev-layer output
    const float2* __restrict__ stIn,   // [I] prev-layer (scale, shift)
    half_t* __restrict__ Y,            // [P][O] f16 raw output
    float* __restrict__ sums,          // [O]
    float* __restrict__ sumsq,         // [O]
    const float* __restrict__ gbias,   // [B][O] optional per-(batch,o) bias
    int O, int I)
{
    __shared__ __align__(16) half_t Xt[128][34];   // 17-dword stride: no conflicts
    __shared__ __align__(16) half_t Wt[64][40];    // 80B stride: 16B-aligned rows,
                                                   // 20-dword stride conflict-free
    __shared__ float2 stS[1024];
    __shared__ float  sLds[64];
    __shared__ float  qLds[64];

    const int t    = threadIdx.x;
    const int lane = t & 31;
    const int wave = t >> 5;
    const int p0   = blockIdx.x * 128;
    const int o0   = blockIdx.y * 64;

    if (HAS_ST) for (int i = t; i < I; i += 256) stS[i] = stIn[i];
    if (t < 64) { sLds[t] = 0.f; qLds[t] = 0.f; }
    __syncthreads();

    // Per-thread W-tile chunk: 64 rows x 4 x 16B chunks
    const int wrow = t >> 2, wch = (t & 3) * 8;
    // LDS destination byte offset (low 32 bits of the shared-aperture address)
    const uint32_t wLds = (uint32_t)(uintptr_t)(void*)&Wt[wrow][wch];

    v8f acc[4];
#pragma unroll
    for (int ot = 0; ot < 4; ++ot) acc[ot] = (v8f){0.f,0.f,0.f,0.f,0.f,0.f,0.f,0.f};

    for (int k0 = 0; k0 < I; k0 += 32) {
        // ---- stage W tile (64 x 32) asynchronously: global -> LDS ----
        {
            uint64_t gaddr = (uint64_t)(uintptr_t)(W + (size_t)(o0 + wrow) * I + k0 + wch);
            asm volatile("global_load_async_to_lds_b128 %0, %1, off"
                         :: "v"(wLds), "v"(gaddr) : "memory");
        }
        // ---- stage X tile (128 x 32), fold prev-layer BN + ReLU ----
#pragma unroll
        for (int rep = 0; rep < 2; ++rep) {
            int idx = t + rep * 256;
            int row = idx >> 2, ch = (idx & 3) * 8;
            union { uint4 q; half_t h[8]; uint32_t u[4]; } u;
            u.q = *(const uint4*)(X + (size_t)(p0 + row) * I + k0 + ch);
            if (HAS_ST) {
#pragma unroll
                for (int m = 0; m < 8; ++m) {
                    float2 s = stS[k0 + ch + m];
                    u.h[m] = (half_t)fmaxf((float)u.h[m] * s.x + s.y, 0.f);
                }
            }
#pragma unroll
            for (int m = 0; m < 4; ++m)
                *(uint32_t*)&Xt[row][ch + 2 * m] = u.u[m];
        }
        if (k0 + 32 < I)  // hint next K-chunk into cache (global_prefetch_b8)
            __builtin_prefetch(X + (size_t)(p0 + (t >> 1)) * I + k0 + 32, 0, 1);
        asm volatile("s_wait_asynccnt 0x0" ::: "memory");   // W tile landed in LDS
        __syncthreads();

        // ---- A fragment: 16 points x 32 K (doc 16-bit A layout) ----
        const int mrow  = wave * 16 + (lane & 15);
        const int khalf = lane >> 4;
        v16h a;
#pragma unroll
        for (int v = 0; v < 8; ++v) {
            int k = ((v < 4) ? 0 : 16) + khalf * 8 + (v & 3) * 2;
            a[2 * v]     = Xt[mrow][k];
            a[2 * v + 1] = Xt[mrow][k + 1];
        }
        // ---- 4 B fragments (16 out-ch each) + WMMA ----
#pragma unroll
        for (int ot = 0; ot < 4; ++ot) {
            const int col = ot * 16 + (lane & 15);
            const int kb  = khalf * 8;
            v16h bf;
#pragma unroll
            for (int v = 0; v < 8; ++v) {
                int k = ((v < 4) ? 0 : 16) + kb + (v & 3) * 2;
                bf[2 * v]     = Wt[col][k];
                bf[2 * v + 1] = Wt[col][k + 1];
            }
            acc[ot] = __builtin_amdgcn_wmma_f32_16x16x32_f16(
                false, a, false, bf, (short)0, acc[ot], false, false);
        }
        __syncthreads();
    }

    // ---- epilogue: store raw f16, accumulate BN stats ----
    const int batch = p0 / NPTS;   // 128 | NPTS, uniform per block
#pragma unroll
    for (int ot = 0; ot < 4; ++ot) {
        const int o    = o0 + ot * 16 + (lane & 15);
        const float gb = HAS_GB ? gbias[batch * O + o] : 0.f;
        float s = 0.f, q = 0.f;
#pragma unroll
        for (int j = 0; j < 8; ++j) {
            float v = acc[ot][j] + gb;
            int p = p0 + wave * 16 + (lane >> 4) * 8 + j;
            Y[(size_t)p * O + o] = (half_t)v;
            s += v; q += v * v;
        }
        unsafeAtomicAdd(&sLds[ot * 16 + (lane & 15)], s);
        unsafeAtomicAdd(&qLds[ot * 16 + (lane & 15)], q);
    }
    __syncthreads();
    if (t < 64) {
        unsafeAtomicAdd(&sums[o0 + t],  sLds[t]);
        unsafeAtomicAdd(&sumsq[o0 + t], qLds[t]);
    }
}

// Per-channel (scale, shift) = fold of BN affine
__global__ void bn_finalize(const float* __restrict__ sums,
                            const float* __restrict__ sumsq,
                            const float* __restrict__ g,
                            const float* __restrict__ bta,
                            float2* __restrict__ st, int O)
{
    int o = blockIdx.x * 256 + threadIdx.x;
    if (o >= O) return;
    const float invP = 1.f / (float)PTOT;
    float m  = sums[o] * invP;
    float v  = fmaxf(sumsq[o] * invP - m * m, 0.f);
    float sc = g[o] * rsqrtf(v + 1e-5f);
    st[o] = make_float2(sc, bta[o] - m * sc);
}

// Global max-pool of relu(bn(Y3)) over points, per (batch, channel)
__global__ __launch_bounds__(256) void maxpool_relu(
    const half_t* __restrict__ Y3,   // [P][1024]
    const float2* __restrict__ st3,  // [1024]
    float* __restrict__ fglob)       // [B][1024]
{
    int b  = blockIdx.x;
    int og = blockIdx.y * 64;
    int t  = threadIdx.x;
    int oc = t & 63, pp = t >> 6;
    float2 s = st3[og + oc];
    float m = -3.4e38f;
    for (int n = pp; n < NPTS; n += 4) {
        float z = (float)Y3[(size_t)(b * NPTS + n) * 1024 + og + oc];
        m = fmaxf(m, z * s.x + s.y);
    }
    __shared__ float red[256];
    red[t] = m;
    __syncthreads();
    if (pp == 0) {
        m = fmaxf(fmaxf(red[oc], red[64 + oc]), fmaxf(red[128 + oc], red[192 + oc]));
        fglob[b * 1024 + og + oc] = fmaxf(m, 0.f);
    }
}

// gbias[b][o] = sum_i dw0[o][1024+i] * fglob[b][i]  (one warp per (b,o))
__global__ __launch_bounds__(256) void glob_bias(
    const float* __restrict__ dw0,    // [512][2048]
    const float* __restrict__ fglob,  // [B][1024]
    float* __restrict__ gbias)        // [B][512]
{
    int wid  = blockIdx.x * 8 + (threadIdx.x >> 5);
    int lane = threadIdx.x & 31;
    int b = wid >> 9, o = wid & 511;
    float s = 0.f;
    for (int i = lane; i < 1024; i += 32)
        s += dw0[(size_t)o * 2048 + 1024 + i] * fglob[b * 1024 + i];
#pragma unroll
    for (int off = 16; off > 0; off >>= 1) s += __shfl_down(s, off, 32);
    if (lane == 0) gbias[b * 512 + o] = s;
}

// Final 128->16 layer (+bias, no BN) + softmax over 16 clusters
__global__ __launch_bounds__(256) void dec3_softmax(
    const half_t* __restrict__ X2,    // [P][128] raw D2 output
    const float2* __restrict__ st2,   // [128]
    const float* __restrict__ dw3,    // [16][128]
    const float* __restrict__ db3,    // [16]
    float* __restrict__ gamma)        // [16][P]
{
    __shared__ float  wsh[16 * 128];
    __shared__ float  bsh[16];
    __shared__ float2 stsh[128];
    int t = threadIdx.x;
    for (int i = t; i < 2048; i += 256) wsh[i] = dw3[i];
    if (t < 16)  bsh[t]  = db3[t];
    if (t < 128) stsh[t] = st2[t];
    __syncthreads();

    int p = blockIdx.x * 256 + t;
    float acc[16];
#pragma unroll
    for (int j = 0; j < 16; ++j) acc[j] = bsh[j];
    const half_t* xp = X2 + (size_t)p * 128;
    for (int i = 0; i < 128; ++i) {
        float2 s = stsh[i];
        float xv = fmaxf((float)xp[i] * s.x + s.y, 0.f);
#pragma unroll
        for (int j = 0; j < 16; ++j) acc[j] += wsh[j * 128 + i] * xv;
    }
    float m = acc[0];
#pragma unroll
    for (int j = 1; j < 16; ++j) m = fmaxf(m, acc[j]);
    float sum = 0.f;
#pragma unroll
    for (int j = 0; j < 16; ++j) { acc[j] = __expf(acc[j] - m); sum += acc[j]; }
    float inv = 1.f / sum;
#pragma unroll
    for (int j = 0; j < 16; ++j)
        gamma[(size_t)j * PTOT + p] = acc[j] * inv;
}

// GMM moments per (batch, cluster): pi, mu, isotropic variance s
__global__ __launch_bounds__(256) void gmm_stats(
    const float* __restrict__ gamma,  // [16][P]
    const float* __restrict__ inp,    // [B][N][83]
    float* __restrict__ pi, float* __restrict__ mu, float* __restrict__ sct)
{
    int b = blockIdx.x >> 4, j = blockIdx.x & 15;
    int t = threadIdx.x;
    float S0 = 0, Sx = 0, Sy = 0, Sz = 0, S2 = 0;
    for (int n = t; n < NPTS; n += 256) {
        float g = gamma[(size_t)j * PTOT + b * NPTS + n];
        const float* pt = inp + ((size_t)b * NPTS + n) * FEAT_STRIDE;
        float x = pt[0], y = pt[1], z = pt[2];
        S0 += g; Sx += g * x; Sy += g * y; Sz += g * z;
        S2 += g * (x * x + y * y + z * z);
    }
    __shared__ float red[5][256];
    red[0][t] = S0; red[1][t] = Sx; red[2][t] = Sy; red[3][t] = Sz; red[4][t] = S2;
    __syncthreads();
    for (int st = 128; st > 0; st >>= 1) {
        if (t < st)
#pragma unroll
            for (int k = 0; k < 5; ++k) red[k][t] += red[k][t + st];
        __syncthreads();
    }
    if (t == 0) {
        S0 = red[0][0]; Sx = red[1][0]; Sy = red[2][0]; Sz = red[3][0]; S2 = red[4][0];
        int bj = b * 16 + j;
        float inv = 1.f / S0;
        pi[bj] = S0 / (float)NPTS;
        mu[bj * 3 + 0] = Sx * inv;
        mu[bj * 3 + 1] = Sy * inv;
        mu[bj * 3 + 2] = Sz * inv;
        sct[bj] = (S2 - (Sx * Sx + Sy * Sy + Sz * Sz) * inv) * inv;
    }
}

// ---------------------------------------------------------------------------
// Weighted Kabsch with isotropic target covariances (3x3 Jacobi SVD)
// ---------------------------------------------------------------------------
__device__ inline float det3(const float V[3][3])
{
    return V[0][0] * (V[1][1] * V[2][2] - V[1][2] * V[2][1])
         - V[0][1] * (V[1][0] * V[2][2] - V[1][2] * V[2][0])
         + V[0][2] * (V[1][0] * V[2][1] - V[1][1] * V[2][0]);
}

__device__ void kabsch(const float* pi, const float* muS, const float* muT,
                       const float* sigT, float R[3][3], float tr[3])
{
    float cs[3] = {0, 0, 0}, ct[3] = {0, 0, 0};
    for (int j = 0; j < 16; ++j)
        for (int d = 0; d < 3; ++d) {
            cs[d] += pi[j] * muS[j * 3 + d];
            ct[d] += pi[j] * muT[j * 3 + d];
        }
    float M[3][3] = {};
    for (int j = 0; j < 16; ++j) {
        float w  = pi[j] / fmaxf(sigT[j], 1e-12f);  // inv of isotropic sigma
        float a0 = muS[j * 3 + 0] - cs[0], a1 = muS[j * 3 + 1] - cs[1], a2 = muS[j * 3 + 2] - cs[2];
        float b0 = muT[j * 3 + 0] - ct[0], b1 = muT[j * 3 + 1] - ct[1], b2 = muT[j * 3 + 2] - ct[2];
        M[0][0] += w * a0 * b0; M[0][1] += w * a0 * b1; M[0][2] += w * a0 * b2;
        M[1][0] += w * a1 * b0; M[1][1] += w * a1 * b1; M[1][2] += w * a1 * b2;
        M[2][0] += w * a2 * b0; M[2][1] += w * a2 * b1; M[2][2] += w * a2 * b2;
    }
    // B = M^T M, Jacobi eigen -> right singular vectors V, eigenvalues
    float Bm[3][3];
    for (int r = 0; r < 3; ++r)
        for (int c = 0; c < 3; ++c)
            Bm[r][c] = M[0][r] * M[0][c] + M[1][r] * M[1][c] + M[2][r] * M[2][c];
    float V[3][3] = {{1, 0, 0}, {0, 1, 0}, {0, 0, 1}};
    const int JP[3] = {0, 0, 1}, JQ[3] = {1, 2, 2};
    for (int sweep = 0; sweep < 8; ++sweep)
        for (int r = 0; r < 3; ++r) {
            int p = JP[r], q = JQ[r];
            float apq = Bm[p][q];
            if (fabsf(apq) < 1e-20f) continue;
            float theta = (Bm[q][q] - Bm[p][p]) / (2.f * apq);
            float tt = copysignf(1.f, theta) / (fabsf(theta) + sqrtf(theta * theta + 1.f));
            float c  = rsqrtf(tt * tt + 1.f);
            float sn = tt * c;
            for (int k = 0; k < 3; ++k) {
                float akp = Bm[k][p], akq = Bm[k][q];
                Bm[k][p] = c * akp - sn * akq;
                Bm[k][q] = sn * akp + c * akq;
            }
            for (int k = 0; k < 3; ++k) {
                float apk = Bm[p][k], aqk = Bm[q][k];
                Bm[p][k] = c * apk - sn * aqk;
                Bm[q][k] = sn * apk + c * aqk;
            }
            for (int k = 0; k < 3; ++k) {
                float vkp = V[k][p], vkq = V[k][q];
                V[k][p] = c * vkp - sn * vkq;
                V[k][q] = sn * vkp + c * vkq;
            }
        }
    float lam[3] = {Bm[0][0], Bm[1][1], Bm[2][2]};
    for (int i = 0; i < 2; ++i)
        for (int j = i + 1; j < 3; ++j)
            if (lam[j] > lam[i]) {
                float tl = lam[i]; lam[i] = lam[j]; lam[j] = tl;
                for (int k = 0; k < 3; ++k) { float tv = V[k][i]; V[k][i] = V[k][j]; V[k][j] = tv; }
            }
    // Left singular vectors: u_i = M v_i / |.|, Gram-Schmidt, u2 = u0 x u1
    float U[3][3];
    float u[3];
    for (int r = 0; r < 3; ++r)
        u[r] = M[r][0] * V[0][0] + M[r][1] * V[1][0] + M[r][2] * V[2][0];
    float n0 = sqrtf(u[0] * u[0] + u[1] * u[1] + u[2] * u[2]);
    if (n0 > 1e-18f) { u[0] /= n0; u[1] /= n0; u[2] /= n0; }
    else             { u[0] = 1.f; u[1] = 0.f; u[2] = 0.f; }
    U[0][0] = u[0]; U[1][0] = u[1]; U[2][0] = u[2];
    for (int r = 0; r < 3; ++r)
        u[r] = M[r][0] * V[0][1] + M[r][1] * V[1][1] + M[r][2] * V[2][1];
    float d01 = u[0] * U[0][0] + u[1] * U[1][0] + u[2] * U[2][0];
    u[0] -= d01 * U[0][0]; u[1] -= d01 * U[1][0]; u[2] -= d01 * U[2][0];
    float n1 = sqrtf(u[0] * u[0] + u[1] * u[1] + u[2] * u[2]);
    if (n1 > 1e-18f) { u[0] /= n1; u[1] /= n1; u[2] /= n1; }
    else {
        u[0] = -U[1][0]; u[1] = U[0][0]; u[2] = 0.f;   // any orthogonal vector
        float nn = sqrtf(u[0] * u[0] + u[1] * u[1]);
        if (nn > 1e-18f) { u[0] /= nn; u[1] /= nn; } else { u[0] = 0.f; u[1] = 0.f; u[2] = 1.f; }
    }
    U[0][1] = u[0]; U[1][1] = u[1]; U[2][1] = u[2];
    U[0][2] = U[1][0] * U[2][1] - U[2][0] * U[1][1];
    U[1][2] = U[2][0] * U[0][1] - U[0][0] * U[2][1];
    U[2][2] = U[0][0] * U[1][1] - U[1][0] * U[0][1];

    float d = det3(V) * det3(U);     // det(V U^T)
    for (int a = 0; a < 3; ++a)
        for (int c = 0; c < 3; ++c)
            R[a][c] = V[a][0] * U[c][0] + V[a][1] * U[c][1] + d * V[a][2] * U[c][2];
    for (int a = 0; a < 3; ++a)
        tr[a] = ct[a] - (R[a][0] * cs[0] + R[a][1] * cs[1] + R[a][2] * cs[2]);
}

__global__ void register_out(const float* __restrict__ piT, const float* __restrict__ muT,
                             const float* __restrict__ sT,
                             const float* __restrict__ piS, const float* __restrict__ muS,
                             const float* __restrict__ sS,
                             float* __restrict__ Rt, float* __restrict__ out)
{
    int b = threadIdx.x;
    if (b >= BB) return;
    float Re[3][3], te[3], Ri[3][3], ti[3];
    // R_est = register(s_pi, s_mu, t_mu, t_sigma); R_inv = register(t_pi, t_mu, s_mu, s_sigma)
    kabsch(piS + b * 16, muS + b * 48, muT + b * 48, sT + b * 16, Re, te);
    kabsch(piT + b * 16, muT + b * 48, muS + b * 48, sS + b * 16, Ri, ti);
    float* rw = Rt + b * 12;
    for (int r = 0; r < 3; ++r) {
        for (int c = 0; c < 3; ++c) rw[r * 3 + c] = Re[r][c];
        rw[9 + r] = te[r];
    }
    float* oe = out + (size_t)b * OUT_ROW + NPTS * 3;
    for (int r = 0; r < 3; ++r) {
        oe[r * 4 + 0] = Re[r][0]; oe[r * 4 + 1] = Re[r][1];
        oe[r * 4 + 2] = Re[r][2]; oe[r * 4 + 3] = te[r];
    }
    oe[12] = 0.f; oe[13] = 0.f; oe[14] = 0.f; oe[15] = 1.f;
    float* oi = oe + 16;
    for (int r = 0; r < 3; ++r) {
        oi[r * 4 + 0] = Ri[r][0]; oi[r * 4 + 1] = Ri[r][1];
        oi[r * 4 + 2] = Ri[r][2]; oi[r * 4 + 3] = ti[r];
    }
    oi[12] = 0.f; oi[13] = 0.f; oi[14] = 0.f; oi[15] = 1.f;
}

__global__ __launch_bounds__(256) void transform_pts(
    const float* __restrict__ src, const float* __restrict__ Rt, float* __restrict__ out)
{
    int idx = blockIdx.x * 256 + threadIdx.x;
    if (idx >= PTOT) return;
    int b = idx >> 14, n = idx & (NPTS - 1);
    const float* R = Rt + b * 12;
    const float* p = src + (size_t)idx * FEAT_STRIDE;
    float x = p[0], y = p[1], z = p[2];
    float* o = out + (size_t)b * OUT_ROW + n * 3;
    o[0] = R[0] * x + R[1] * y + R[2] * z + R[9];
    o[1] = R[3] * x + R[4] * y + R[5] * z + R[10];
    o[2] = R[6] * x + R[7] * y + R[8] * z + R[11];
}

// ---------------------------------------------------------------------------
extern "C" void kernel_launch(void* const* d_in, const int* in_sizes, int n_in,
                              void* d_out, int out_size, void* d_ws, size_t ws_size,
                              hipStream_t stream)
{
    (void)in_sizes; (void)n_in; (void)out_size; (void)ws_size;

    const float* tmpl = (const float*)d_in[0];
    const float* srcp = (const float*)d_in[1];
    const float* ew0 = (const float*)d_in[2];
    const float* eg0 = (const float*)d_in[3];
    const float* eb0 = (const float*)d_in[4];
    const float* ew1 = (const float*)d_in[5];
    const float* eg1 = (const float*)d_in[6];
    const float* eb1 = (const float*)d_in[7];
    const float* ew2 = (const float*)d_in[8];
    const float* eg2 = (const float*)d_in[9];
    const float* eb2 = (const float*)d_in[10];
    const float* ew3 = (const float*)d_in[11];
    const float* eg3 = (const float*)d_in[12];
    const float* eb3 = (const float*)d_in[13];
    const float* dw0 = (const float*)d_in[14];
    const float* dg0 = (const float*)d_in[15];
    const float* db0 = (const float*)d_in[16];
    const float* dw1 = (const float*)d_in[17];
    const float* dg1 = (const float*)d_in[18];
    const float* db1 = (const float*)d_in[19];
    const float* dw2 = (const float*)d_in[20];
    const float* dg2 = (const float*)d_in[21];
    const float* db2 = (const float*)d_in[22];
    const float* dw3 = (const float*)d_in[23];
    const float* db3 = (const float*)d_in[24];
    float* out = (float*)d_out;

    char*  ws  = (char*)d_ws;
    size_t off = 0;
    auto alloc = [&](size_t bytes) -> void* {
        void* p = ws + off;
        off += (bytes + 255) & ~(size_t)255;
        return p;
    };
    half_t* buf0 = (half_t*)alloc((size_t)PTOT * 1024 * sizeof(half_t));
    half_t* buf1 = (half_t*)alloc((size_t)PTOT * 1024 * sizeof(half_t));
    half_t* wE0 = (half_t*)alloc((size_t)64 * 96 * 2);
    half_t* wE1 = (half_t*)alloc((size_t)128 * 64 * 2);
    half_t* wE2 = (half_t*)alloc((size_t)256 * 128 * 2);
    half_t* wE3 = (half_t*)alloc((size_t)1024 * 256 * 2);
    half_t* wD0 = (half_t*)alloc((size_t)512 * 1024 * 2);
    half_t* wD1 = (half_t*)alloc((size_t)256 * 512 * 2);
    half_t* wD2 = (half_t*)alloc((size_t)128 * 256 * 2);
    float*  stats  = (float*)alloc(2048 * 4);    // sums[1024] | sumsq[1024]
    float*  sums   = stats;
    float*  sumsq  = stats + 1024;
    float2* stA    = (float2*)alloc(1024 * 8);
    float2* stB    = (float2*)alloc(1024 * 8);
    float*  fglob  = (float*)alloc(4 * 1024 * 4);
    float*  gbias  = (float*)alloc(4 * 512 * 4);
    float*  gammaT = (float*)alloc((size_t)16 * PTOT * 4);
    float*  gammaS = (float*)alloc((size_t)16 * PTOT * 4);
    float*  piT = (float*)alloc(64 * 4);
    float*  muT = (float*)alloc(192 * 4);
    float*  sT  = (float*)alloc(64 * 4);
    float*  piS = (float*)alloc(64 * 4);
    float*  muS = (float*)alloc(192 * 4);
    float*  sS  = (float*)alloc(64 * 4);
    float*  Rt  = (float*)alloc(48 * 4);

    // ---- convert weights to f16 (ew0 zero-padded 80 -> 96 in K) ----
    auto cvt = [&](const float* s, half_t* dptr, int O, int I, int Ipad, int stride) {
        int n = O * Ipad;
        cvt_w<<<(n + 255) / 256, 256, 0, stream>>>(s, dptr, O, I, Ipad, stride);
    };
    cvt(ew0, wE0, 64, 80, 96, 80);
    cvt(ew1, wE1, 128, 64, 64, 64);
    cvt(ew2, wE2, 256, 128, 128, 128);
    cvt(ew3, wE3, 1024, 256, 256, 256);
    cvt(dw0, wD0, 512, 1024, 1024, 2048);   // local half of decoder-0 weights
    cvt(dw1, wD1, 256, 512, 512, 512);
    cvt(dw2, wD2, 128, 256, 256, 256);

    auto run_gemm = [&](const half_t* Wp, const half_t* Xp, const float2* stIn,
                        half_t* Yp, const float* gb, int O, int I, int mode) {
        zero_f<<<(2048 + 255) / 256, 256, 0, stream>>>(stats, 2048);
        dim3 g(PTOT / 128, O / 64);
        if (mode == 0)
            gemm_bn_wmma<false, false><<<g, 256, 0, stream>>>(Wp, Xp, nullptr, Yp, sums, sumsq, nullptr, O, I);
        else if (mode == 1)
            gemm_bn_wmma<true, false><<<g, 256, 0, stream>>>(Wp, Xp, stIn, Yp, sums, sumsq, nullptr, O, I);
        else
            gemm_bn_wmma<true, true><<<g, 256, 0, stream>>>(Wp, Xp, stIn, Yp, sums, sumsq, gb, O, I);
    };
    auto fin = [&](const float* g, const float* b, float2* st, int O) {
        bn_finalize<<<(O + 255) / 256, 256, 0, stream>>>(sums, sumsq, g, b, st, O);
    };

    auto backbone = [&](const float* inp, float* gamma) {
        pack_input<<<(PTOT * DINP + 255) / 256, 256, 0, stream>>>(inp, buf0);
        run_gemm(wE0, buf0, nullptr, buf1, nullptr, 64, 96, 0);    fin(eg0, eb0, stA, 64);
        run_gemm(wE1, buf1, stA, buf0, nullptr, 128, 64, 1);       fin(eg1, eb1, stB, 128);
        run_gemm(wE2, buf0, stB, buf1, nullptr, 256, 128, 1);      fin(eg2, eb2, stA, 256);
        run_gemm(wE3, buf1, stA, buf0, nullptr, 1024, 256, 1);     fin(eg3, eb3, stB, 1024);
        maxpool_relu<<<dim3(BB, 16), 256, 0, stream>>>(buf0, stB, fglob);
        glob_bias<<<256, 256, 0, stream>>>(dw0, fglob, gbias);
        run_gemm(wD0, buf0, stB, buf1, gbias, 512, 1024, 2);       fin(dg0, db0, stA, 512);
        run_gemm(wD1, buf1, stA, buf0, nullptr, 256, 512, 1);      fin(dg1, db1, stB, 256);
        run_gemm(wD2, buf0, stB, buf1, nullptr, 128, 256, 1);      fin(dg2, db2, stA, 128);
        dec3_softmax<<<PTOT / 256, 256, 0, stream>>>(buf1, stA, dw3, db3, gamma);
    };

    backbone(tmpl, gammaT);
    gmm_stats<<<BB * 16, 256, 0, stream>>>(gammaT, tmpl, piT, muT, sT);
    backbone(srcp, gammaS);
    gmm_stats<<<BB * 16, 256, 0, stream>>>(gammaS, srcp, piS, muS, sS);

    register_out<<<1, 32, 0, stream>>>(piT, muT, sT, piS, muS, sS, Rt, out);
    transform_pts<<<PTOT / 256, 256, 0, stream>>>(srcp, Rt, out);
}
